// WindowAttention3D_4947802325773
// MI455X (gfx1250) — matrix operand order repaired and checked by reading, not compiled, not run on hardware
//
#include <hip/hip_runtime.h>
#include <hip/hip_bf16.h>

// ---------------------------------------------------------------------------
// Fused 3D window attention for MI455X (gfx1250), wave32 + WMMA bf16.
//   window = (2,7,7) -> N=98 tokens, DIM=128, HEADS=4 (hd=32), B_=2048, nW=64
// One workgroup (256 threads = 8 waves) handles one window entirely in LDS.
// 157.4 KB LDS -> 2 workgroups per 320 KB WGP.
// ---------------------------------------------------------------------------

typedef __attribute__((ext_vector_type(16))) __bf16 v16bf;
typedef __attribute__((ext_vector_type(8)))  float  v8f;

#define NTHREADS   256
#define NTOK       98
#define NPAD       112          // 7 tiles of 16 rows
#define DIMC       128
#define HEADS4     4
#define SX_STRIDE  136          // 128 cols + 8 pad (16B-aligned, conflict-free)
#define QK_STRIDE  32           // tight (4-way conflict on frag loads, saves 14KB)
#define VT_STRIDE  136          // 128 token cols + 8 pad
#define P_STRIDE   136          // 128 cols + 8 pad

static __device__ __forceinline__ unsigned short f2bf(float f) {
    union { float f; unsigned u; } v; v.f = f;
    unsigned r = v.u + 0x7fffu + ((v.u >> 16) & 1u);   // round-to-nearest-even
    return (unsigned short)(r >> 16);
}

// Build a 16-element bf16 fragment from two 16-byte chunks (LDS or global).
static __device__ __forceinline__ v16bf load_frag2(const unsigned short* p0,
                                                   const unsigned short* p1) {
    union { v16bf v; uint4 u[2]; } f;
    f.u[0] = *(const uint4*)p0;
    f.u[1] = *(const uint4*)p1;
    return f.v;
}

// --------------------------- prep: fp32 weights -> bf16 in workspace --------
__global__ void wconv_kernel(const float* __restrict__ qkv_w,
                             const float* __restrict__ proj_w,
                             unsigned short* __restrict__ wq,
                             unsigned short* __restrict__ wp) {
    int i = blockIdx.x * blockDim.x + threadIdx.x;
    if (i < 384 * 128) wq[i] = f2bf(qkv_w[i]);
    if (i < 128 * 128) wp[i] = f2bf(proj_w[i]);
}

// --------------------------- fused attention --------------------------------
__global__ __launch_bounds__(NTHREADS, 2)
void swin_attn_fused(const float* __restrict__ x,
                     const float* __restrict__ mask,
                     const float* __restrict__ qkv_b,
                     const float* __restrict__ rpb,      // [507,4]
                     const float* __restrict__ proj_b,
                     const unsigned short* __restrict__ wq,   // [384,128] bf16
                     const unsigned short* __restrict__ wp,   // [128,128] bf16
                     float* __restrict__ out) {
    __shared__ unsigned short s_x [NPAD * SX_STRIDE];            // x, then attn-out
    __shared__ unsigned short s_q [HEADS4 * NPAD * QK_STRIDE];   // q (pre-scaled)
    __shared__ unsigned short s_k [HEADS4 * NPAD * QK_STRIDE];
    __shared__ unsigned short s_vT[HEADS4 * 32 * VT_STRIDE];     // v transposed [hd][tok]
    __shared__ unsigned short s_p [8 * 16 * P_STRIDE];           // per-wave softmax probs

    const int tid   = threadIdx.x;
    // wave index is wave-uniform: pin it to an SGPR so all loop control /
    // tile indices / q-k-v branches become SALU instead of exec-mask VALU.
    const int wave  = __builtin_amdgcn_readfirstlane(tid >> 5);
    const int lane  = tid & 31;
    const int lhalf = lane >> 4;   // 0/1 (WMMA half selector)
    const int lcol  = lane & 15;   // N-column / row selector
    const int b     = blockIdx.x;

    // ---- zero all LDS (pad rows / pad cols must be 0.0 for WMMA K-padding)
    {
        uint4 z = make_uint4(0u, 0u, 0u, 0u);
        for (int i = tid; i < (NPAD * SX_STRIDE) / 8; i += NTHREADS)          ((uint4*)s_x)[i]  = z;
        for (int i = tid; i < (HEADS4 * NPAD * QK_STRIDE) / 8; i += NTHREADS) ((uint4*)s_q)[i]  = z;
        for (int i = tid; i < (HEADS4 * NPAD * QK_STRIDE) / 8; i += NTHREADS) ((uint4*)s_k)[i]  = z;
        for (int i = tid; i < (HEADS4 * 32 * VT_STRIDE) / 8; i += NTHREADS)   ((uint4*)s_vT)[i] = z;
        for (int i = tid; i < (8 * 16 * P_STRIDE) / 8; i += NTHREADS)         ((uint4*)s_p)[i]  = z;
    }
    __syncthreads();

    // ---- phase 1: x -> bf16 LDS (vectorized: float4 in, uint2 out) --------
    const float* xb = x + (size_t)b * (NTOK * DIMC);
    for (int i = tid; i < (NTOK * DIMC) / 4; i += NTHREADS) {
        int e = i << 2;
        int r = e >> 7, c = e & 127;
        float4 f = ((const float4*)xb)[i];
        uint2 pk;
        pk.x = (unsigned)f2bf(f.x) | ((unsigned)f2bf(f.y) << 16);
        pk.y = (unsigned)f2bf(f.z) | ((unsigned)f2bf(f.w) << 16);
        *(uint2*)&s_x[r * SX_STRIDE + c] = pk;
    }
    __syncthreads();

    // ---- phase 2: QKV GEMM  [112,128] x [128,384] ------------------------
    // 7 row-tiles x 24 col-tiles, K = 128 in 4 WMMA k-steps.
    for (int t = wave; t < 7 * 24; t += 8) {
        int mt = t / 24, nt = t % 24;            // scalar
        int n  = nt * 16 + lcol;                 // global output column
        v8f acc = {};
        for (int ks = 0; ks < 4; ++ks) {
            int k0    = ks * 32;
            int arow  = mt * 16 + lcol;
            int abase = k0 + lhalf * 8;          // A layout: runs [b..b+7],[b+16..b+23]
            v16bf af = load_frag2(&s_x[arow * SX_STRIDE + abase],
                                  &s_x[arow * SX_STRIDE + abase + 16]);
            const unsigned short* bp = wq + n * 128 + k0 + lhalf * 8;   // B: runs [b..b+7],[b+16..b+23]
            v16bf bf = load_frag2(bp, bp + 16);
            acc = __builtin_amdgcn_wmma_f32_16x16x32_bf16(false, af, false, bf,
                                                          (short)0, acc, false, false);
        }
        float bias = qkv_b[n];
        // scalar: a 16-col tile never crosses a q/k/v or head boundary
        int which = nt >> 3;               // 0=q 1=k 2=v
        int head  = (nt >> 1) & 3;
        int hd    = ((nt & 1) << 4) + lcol;
        if (which == 0) {
            for (int j = 0; j < 8; ++j) {
                int row = mt * 16 + j + lhalf * 8;     // C layout: M = j + 8*half
                s_q[(head * NPAD + row) * QK_STRIDE + hd] =
                    f2bf((acc[j] + bias) * 0.17677669529663687f);
            }
        } else if (which == 1) {
            for (int j = 0; j < 8; ++j) {
                int row = mt * 16 + j + lhalf * 8;
                s_k[(head * NPAD + row) * QK_STRIDE + hd] = f2bf(acc[j] + bias);
            }
        } else {
            for (int j = 0; j < 8; ++j) {
                int row = mt * 16 + j + lhalf * 8;
                s_vT[(head * 32 + hd) * VT_STRIDE + row] = f2bf(acc[j] + bias);
            }
        }
    }
    __syncthreads();

    // ---- phase 3: attention, one (head,row-tile) task per wave iteration --
    const float* mrow = mask + (size_t)(b & 63) * (NTOK * NTOK);
    for (int task = wave; task < 28; task += 8) {
        int head = task & 3, mt = task >> 2;     // scalar

        // q A-fragment (rows fixed for the whole task, hd = K = 32)
        int qrow = mt * 16 + lcol;
        int qb   = lhalf * 8;
        const unsigned short* qp = &s_q[(head * NPAD + qrow) * QK_STRIDE + qb];
        v16bf qf = load_frag2(qp, qp + 16);

        // scores: 7 col-tiles, one WMMA each (K = hd = 32)
        v8f sc[7];
        for (int ct = 0; ct < 7; ++ct) {
            int ktok = ct * 16 + lcol;
            int kk   = lhalf * 8;
            const unsigned short* kp = &s_k[(head * NPAD + ktok) * QK_STRIDE + kk];
            v16bf kf = load_frag2(kp, kp + 16);
            v8f z = {};
            sc[ct] = __builtin_amdgcn_wmma_f32_16x16x32_bf16(false, qf, false, kf,
                                                             (short)0, z, false, false);
        }

        // rel-pos bias base per row:  idx = (d1-d2+1)*169+(h1-h2+6)*13+(w1-w2+6)
        //                                  = rbase - cbase + 253
        int   rowb = mt * 16 + lhalf * 8;
        int   rbase4[8];                  // (row linear base + 253) * 4 + head
        bool  rok[8];
        for (int j = 0; j < 8; ++j) {
            int row = rowb + j;
            rok[j] = (row < NTOK);
            int d1 = row / 49, r1 = row - d1 * 49, h1 = r1 / 7, w1 = r1 - h1 * 7;
            rbase4[j] = (d1 * 169 + h1 * 13 + w1 + 253) * 4 + head;
        }

        // bias + mask + bounds masking, then row max
        float rmax[8];
        for (int j = 0; j < 8; ++j) rmax[j] = -3.0e38f;
        for (int ct = 0; ct < 7; ++ct) {
            int col = ct * 16 + lcol;
            int d2 = col / 49, r2 = col - d2 * 49, h2 = r2 / 7, w2 = r2 - h2 * 7;
            int cbase4 = (d2 * 169 + h2 * 13 + w2) * 4;
            bool cok = (col < NTOK);
            for (int j = 0; j < 8; ++j) {
                float s;
                if (rok[j] && cok) {
                    s = sc[ct][j] + rpb[rbase4[j] - cbase4]
                                  + mrow[(rowb + j) * NTOK + col];
                } else {
                    s = -3.0e38f;
                }
                sc[ct][j] = s;
                rmax[j]   = fmaxf(rmax[j], s);
            }
        }
        // row reductions stay within the 16-lane half that owns the row
        for (int j = 0; j < 8; ++j)
            for (int m = 1; m < 16; m <<= 1)
                rmax[j] = fmaxf(rmax[j], __shfl_xor(rmax[j], m, 32));

        // exp, row sum, probs -> per-wave LDS (bf16)
        float rsum[8];
        for (int j = 0; j < 8; ++j) rsum[j] = 0.f;
        unsigned short* pw = &s_p[wave * 16 * P_STRIDE];
        for (int ct = 0; ct < 7; ++ct) {
            int col = ct * 16 + lcol;
            for (int j = 0; j < 8; ++j) {
                float p = __expf(sc[ct][j] - rmax[j]);
                rsum[j] += p;
                pw[(j + lhalf * 8) * P_STRIDE + col] = f2bf(p);
            }
        }
        for (int j = 0; j < 8; ++j)
            for (int m = 1; m < 16; m <<= 1)
                rsum[j] += __shfl_xor(rsum[j], m, 32);

        // 1/sum once per row (v_rcp_f32, 1 ulp — ample for softmax normalize)
        float rinv[8];
        for (int j = 0; j < 8; ++j) rinv[j] = __builtin_amdgcn_rcpf(rsum[j]);

        // intra-wave LDS store -> cross-lane load ordering
        asm volatile("s_wait_dscnt 0x0" ::: "memory");

        // attn @ v : out [16,32], K padded to 128 (pad cols of p / pad rows of v = 0)
        for (int ct2 = 0; ct2 < 2; ++ct2) {
            v8f oacc = {};
            for (int ks = 0; ks < 4; ++ks) {
                int ab = ks * 32 + lhalf * 8;
                const unsigned short* ap = &pw[lcol * P_STRIDE + ab];
                v16bf af = load_frag2(ap, ap + 16);
                int hdc = ct2 * 16 + lcol;
                int tk  = ks * 32 + lhalf * 8;
                const unsigned short* vp = &s_vT[(head * 32 + hdc) * VT_STRIDE + tk];
                v16bf vf = load_frag2(vp, vp + 16);
                oacc = __builtin_amdgcn_wmma_f32_16x16x32_bf16(false, af, false, vf,
                                                               (short)0, oacc, false, false);
            }
            for (int j = 0; j < 8; ++j) {
                int   row = mt * 16 + j + lhalf * 8;
                float v   = oacc[j] * rinv[j];
                s_x[row * SX_STRIDE + head * 32 + ct2 * 16 + lcol] = f2bf(v);  // reuse s_x
            }
        }
    }
    __syncthreads();

    // ---- phase 4: proj GEMM  [112,128] x [128,128] + bias -> global fp32 --
    float* ob = out + (size_t)b * (NTOK * DIMC);
    for (int t = wave; t < 56; t += 8) {
        int mt = t >> 3, nt = t & 7;             // scalar
        int n  = nt * 16 + lcol;
        v8f acc = {};
        for (int ks = 0; ks < 4; ++ks) {
            int k0    = ks * 32;
            int arow  = mt * 16 + lcol;
            int abase = k0 + lhalf * 8;
            v16bf af = load_frag2(&s_x[arow * SX_STRIDE + abase],
                                  &s_x[arow * SX_STRIDE + abase + 16]);
            const unsigned short* bp = wp + n * 128 + k0 + lhalf * 8;
            v16bf bf = load_frag2(bp, bp + 16);
            acc = __builtin_amdgcn_wmma_f32_16x16x32_bf16(false, af, false, bf,
                                                          (short)0, acc, false, false);
        }
        float pb = proj_b[n];
        if (mt < 6) {
            for (int j = 0; j < 8; ++j) {
                int row = mt * 16 + j + lhalf * 8;
                ob[row * DIMC + n] = acc[j] + pb;
            }
        } else {
            for (int j = 0; j < 8; ++j) {
                int row = 96 + j + lhalf * 8;
                if (row < NTOK) ob[row * DIMC + n] = acc[j] + pb;
            }
        }
    }
}

// ---------------------------------------------------------------------------
extern "C" void kernel_launch(void* const* d_in, const int* in_sizes, int n_in,
                              void* d_out, int out_size, void* d_ws, size_t ws_size,
                              hipStream_t stream) {
    const float* x      = (const float*)d_in[0];   // [2048, 98, 128]
    const float* mask   = (const float*)d_in[1];   // [64, 98, 98]
    const float* qkv_w  = (const float*)d_in[2];   // [384, 128]
    const float* qkv_b  = (const float*)d_in[3];   // [384]
    const float* rpb    = (const float*)d_in[4];   // [507, 4]
    const float* proj_w = (const float*)d_in[5];   // [128, 128]
    const float* proj_b = (const float*)d_in[6];   // [128]

    unsigned short* wq = (unsigned short*)d_ws;        // bf16 qkv_w, 98304 B
    unsigned short* wp = wq + 384 * 128;               // bf16 proj_w, 32768 B

    wconv_kernel<<<(384 * 128 + NTHREADS - 1) / NTHREADS, NTHREADS, 0, stream>>>(
        qkv_w, proj_w, wq, wp);

    swin_attn_fused<<<2048, NTHREADS, 0, stream>>>(
        x, mask, qkv_b, rpb, proj_b, wq, wp, (float*)d_out);
}
